// GuidedAttention_88914412962336
// MI455X (gfx1250) — compile-verified, matrix operand order checked by reading, not run on hardware
//
#include <hip/hip_runtime.h>

// ---------------------------------------------------------------------------
// GuidedAttention for MI455X (gfx1250): bf16 WMMA GEMMs + fp32 score pipeline
// Shapes: b=8, n=1024 tokens, c=512, H=8 heads, hd=64, BH=64 batched heads.
// GEMM: block = 8 waves stacked on M; wave = 32x64 tile (2x4 WMMA);
//       K software-pipelined (unconditional preload, peeled tail, unroll 2).
// ---------------------------------------------------------------------------

typedef __attribute__((ext_vector_type(8)))  float   v8f;
typedef __attribute__((ext_vector_type(8)))  __bf16  v8bf;
typedef __attribute__((ext_vector_type(16))) __bf16  v16bf;

__device__ __forceinline__ unsigned short f2bf(float f) {
    unsigned int u = __float_as_uint(f);
    return (unsigned short)((u + 0x7FFFu + ((u >> 16) & 1u)) >> 16);
}

// ------------------------------ cast f32 -> bf16 ---------------------------
__global__ void cast_f32_bf16(const float* __restrict__ in,
                              unsigned short* __restrict__ out, int n) {
    int i = blockIdx.x * blockDim.x + threadIdx.x;
    if (i < n) out[i] = f2bf(in[i]);
}

// ------------------------- generic batched WMMA GEMM -----------------------
// D[z][M,N] = A[z][M,K] * B[z][N,K]^T (+bias[N]) (+resid[M,N]); fp32 out.
// Fragment layout per CDNA5 ISA 7.12.2 (16-bit, 16x32):
//   lane L: row = L%16, K elems {8h..8h+7, 16+8h..16+8h+7}, h = L/16.
__device__ __forceinline__ v16bf ldfrag(const unsigned short* p, int k0) {
    v8bf lo = *(const v8bf*)(p + k0);
    v8bf hi = *(const v8bf*)(p + k0 + 16);
    return __builtin_shufflevector(lo, hi,
        0,1,2,3,4,5,6,7,8,9,10,11,12,13,14,15);
}

#define WMMA_BF16(A_, B_, C_) \
    __builtin_amdgcn_wmma_f32_16x16x32_bf16(false, (A_), false, (B_), (short)0, (C_), false, false)

__global__ __launch_bounds__(256) void wmma_gemm(
    const unsigned short* __restrict__ A, const unsigned short* __restrict__ B,
    float* __restrict__ C, const float* __restrict__ bias,
    const float* __restrict__ resid,
    int M, int N, int K, long long sA, long long sB, long long sC)
{
    const int lane  = threadIdx.x & 31;
    const int wave  = threadIdx.x >> 5;
    const int tileM = blockIdx.y * 256 + wave * 32;   // wave owns 32 rows
    const int tileN = blockIdx.x * 64;                // block shares 64 cols
    const int h     = lane >> 4;
    const int r16   = lane & 15;

    const unsigned short* A0 =
        A + (size_t)blockIdx.z * sA + (size_t)(tileM + r16) * K + 8 * h;
    const unsigned short* A1 = A0 + (size_t)16 * K;
    const unsigned short* Bp[4];
    Bp[0] = B + (size_t)blockIdx.z * sB + (size_t)(tileN + r16) * K + 8 * h;
    Bp[1] = Bp[0] + (size_t)16 * K;
    Bp[2] = Bp[0] + (size_t)32 * K;
    Bp[3] = Bp[0] + (size_t)48 * K;

    v8f acc[2][4];
    #pragma unroll
    for (int i = 0; i < 2; ++i)
        #pragma unroll
        for (int j = 0; j < 4; ++j) acc[i][j] = (v8f){};

    // prologue: fragments for k0 = 0
    v16bf af[2], bf[4];
    af[0] = ldfrag(A0, 0); af[1] = ldfrag(A1, 0);
    #pragma unroll
    for (int j = 0; j < 4; ++j) bf[j] = ldfrag(Bp[j], 0);

    int k0 = 0;
    #pragma unroll 2
    for (; k0 + 32 < K; k0 += 32) {
        // preload next slab (unconditional; overlaps the 8 WMMAs below)
        v16bf naf[2], nbf[4];
        naf[0] = ldfrag(A0, k0 + 32);
        naf[1] = ldfrag(A1, k0 + 32);
        #pragma unroll
        for (int j = 0; j < 4; ++j) nbf[j] = ldfrag(Bp[j], k0 + 32);
        __builtin_prefetch(A0 + k0 + 64, 0, 3);      // speculative: OOB dropped
        __builtin_prefetch(Bp[0] + k0 + 64, 0, 3);

        #pragma unroll
        for (int i = 0; i < 2; ++i)
            #pragma unroll
            for (int j = 0; j < 4; ++j)
                acc[i][j] = WMMA_BF16(af[i], bf[j], acc[i][j]);

        af[0] = naf[0]; af[1] = naf[1];
        #pragma unroll
        for (int j = 0; j < 4; ++j) bf[j] = nbf[j];
    }
    // peeled tail
    #pragma unroll
    for (int i = 0; i < 2; ++i)
        #pragma unroll
        for (int j = 0; j < 4; ++j)
            acc[i][j] = WMMA_BF16(af[i], bf[j], acc[i][j]);

    float* Cb = C + (size_t)blockIdx.z * sC;
    #pragma unroll
    for (int i = 0; i < 2; ++i) {
        #pragma unroll
        for (int j = 0; j < 4; ++j) {
            const int tm = tileM + i * 16;
            const int gn = tileN + j * 16 + r16;
            const float badd = bias ? bias[gn] : 0.0f;
            #pragma unroll
            for (int r = 0; r < 8; ++r) {
                int gm = tm + 8 * h + r;
                size_t idx = (size_t)gm * N + gn;
                float v = acc[i][j][r] + badd;
                if (resid) v += resid[idx];
                Cb[idx] = v;
            }
        }
    }
}

// ------------------- qkv [b,n,3,H,hd] f32 -> q/k/vT bf16 -------------------
__global__ void repack_qkv(const float* __restrict__ qkv,
                           unsigned short* __restrict__ qb,
                           unsigned short* __restrict__ kb,
                           unsigned short* __restrict__ vtb, float scale) {
    int t = blockIdx.x * blockDim.x + threadIdx.x;   // 2^22 elements
    int d   = t & 63;
    int hh  = (t >> 6) & 7;
    int tok = (t >> 9) & 1023;
    int bb  = t >> 19;
    size_t row = ((size_t)bb * 1024 + tok) * 1536;
    float q = qkv[row +          hh * 64 + d];
    float k = qkv[row +  512   + hh * 64 + d];
    float v = qkv[row + 1024   + hh * 64 + d];
    size_t bh = (size_t)bb * 8 + hh;
    qb[(bh * 1024 + tok) * 64 + d]  = f2bf(q * scale);   // fold hd^-0.5
    kb[(bh * 1024 + tok) * 64 + d]  = f2bf(k);
    vtb[(bh * 64 + d) * 1024 + tok] = f2bf(v);           // transposed (N x K)
}

// --------------- 3x3 conv over [b,H,1024,1024] + BN + sigmoid --------------
__global__ __launch_bounds__(256) void conv_bn_sigmoid(
    const float* __restrict__ S, float* __restrict__ O,
    const float* __restrict__ w,  const float* __restrict__ cb,
    const float* __restrict__ gamma, const float* __restrict__ beta,
    const float* __restrict__ mean,  const float* __restrict__ var)
{
    __shared__ float sw[576];                 // [co][ci][3x3]
    int tid = threadIdx.x;
    for (int i = tid; i < 576; i += 256) sw[i] = w[i];
    __syncthreads();

    int j  = blockIdx.x * 256 + tid;
    int i  = blockIdx.y;
    int bb = blockIdx.z;
    const size_t plane = 1024ull * 1024ull;
    const float* Sb = S + (size_t)bb * 8 * plane;

    float acc[8] = {};
    for (int di = -1; di <= 1; ++di) {
        int ii = i + di;
        if (ii < 0 || ii >= 1024) continue;
        for (int dj = -1; dj <= 1; ++dj) {
            int jj = j + dj;
            if (jj < 0 || jj >= 1024) continue;
            int kk = (di + 1) * 3 + (dj + 1);
            #pragma unroll
            for (int ci = 0; ci < 8; ++ci) {
                float v = Sb[ci * plane + (size_t)ii * 1024 + jj];
                #pragma unroll
                for (int co = 0; co < 8; ++co)
                    acc[co] = fmaf(sw[co * 72 + ci * 9 + kk], v, acc[co]);
            }
        }
    }
    #pragma unroll
    for (int co = 0; co < 8; ++co) {
        float inv = gamma[co] * rsqrtf(var[co] + 1e-5f);
        float y = (acc[co] + cb[co]) * inv + (beta[co] - mean[co] * inv);
        y = 1.0f / (1.0f + __expf(-y));
        O[(size_t)bb * 8 * plane + co * plane + (size_t)i * 1024 + j] = y;
    }
}

// ------------------- row softmax (1024 wide) -> bf16 probs -----------------
__global__ __launch_bounds__(256) void softmax_row(const float* __restrict__ X,
                                                   unsigned short* __restrict__ P) {
    __shared__ float red[256];
    size_t row = blockIdx.x;
    const float* xr = X + row * 1024ull;
    int tid = threadIdx.x;
    float v[4];
    float m = -3.4e38f;
    #pragma unroll
    for (int it = 0; it < 4; ++it) { v[it] = xr[tid + 256 * it]; m = fmaxf(m, v[it]); }
    red[tid] = m; __syncthreads();
    for (int s = 128; s > 0; s >>= 1) {
        if (tid < s) red[tid] = fmaxf(red[tid], red[tid + s]);
        __syncthreads();
    }
    m = red[0]; __syncthreads();
    float sum = 0.f;
    #pragma unroll
    for (int it = 0; it < 4; ++it) { v[it] = __expf(v[it] - m); sum += v[it]; }
    red[tid] = sum; __syncthreads();
    for (int s = 128; s > 0; s >>= 1) {
        if (tid < s) red[tid] += red[tid + s];
        __syncthreads();
    }
    float inv = 1.0f / red[0];
    unsigned short* pr = P + row * 1024ull;
    #pragma unroll
    for (int it = 0; it < 4; ++it) pr[tid + 256 * it] = f2bf(v[it] * inv);
}

// --------------- Z [BH,1024,64] f32 -> [b,1024,512] bf16 -------------------
__global__ void repack_z(const float* __restrict__ Zf,
                         unsigned short* __restrict__ Zb) {
    int t = blockIdx.x * blockDim.x + threadIdx.x;   // 2^22 elements
    int d   = t & 63;
    int hh  = (t >> 6) & 7;
    int tok = (t >> 9) & 1023;
    int bb  = t >> 19;
    float z = Zf[(((size_t)bb * 8 + hh) * 1024 + tok) * 64 + d];
    Zb[((size_t)bb * 1024 + tok) * 512 + hh * 64 + d] = f2bf(z);
}

// ---------------------------------------------------------------------------
extern "C" void kernel_launch(void* const* d_in, const int* in_sizes, int n_in,
                              void* d_out, int out_size, void* d_ws, size_t ws_size,
                              hipStream_t stream) {
    const float* x      = (const float*)d_in[0];   // [8,4,16,16,512] = [8192,512]
    const float* qkv_w  = (const float*)d_in[1];   // [1536,512]
    const float* proj_w = (const float*)d_in[2];   // [512,512]
    const float* proj_b = (const float*)d_in[3];   // [512]
    const float* conv_w = (const float*)d_in[4];   // [8,8,3,3]
    const float* conv_b = (const float*)d_in[5];
    const float* bn_g   = (const float*)d_in[6];
    const float* bn_b   = (const float*)d_in[7];
    const float* bn_m   = (const float*)d_in[8];
    const float* bn_v   = (const float*)d_in[9];
    float* out = (float*)d_out;                    // [8192,512]

    char* ws = (char*)d_ws;
    const size_t OFF_XB    = 0;                                   // 8192*512 bf16
    const size_t OFF_WQKV  = OFF_XB    + 8192ull * 512 * 2;       // 1536*512 bf16
    const size_t OFF_WPROJ = OFF_WQKV  + 1536ull * 512 * 2;       // 512*512 bf16
    const size_t OFF_QKVF  = OFF_WPROJ + 512ull  * 512 * 2;       // 8192*1536 f32
    const size_t OFF_QB    = OFF_QKVF  + 8192ull * 1536 * 4;      // 64*1024*64 bf16
    const size_t OFF_KB    = OFF_QB    + 64ull * 1024 * 64 * 2;
    const size_t OFF_VTB   = OFF_KB    + 64ull * 1024 * 64 * 2;   // transposed V
    const size_t OFF_S     = OFF_VTB   + 64ull * 64 * 1024 * 2;   // region1: 256MB f32
    const size_t OFF_SC    = OFF_S     + 64ull * 1024 * 1024 * 4; // region2: 256MB f32
    const size_t OFF_P     = OFF_S;                               // reuse r1 as bf16 P
    const size_t OFF_ZF    = OFF_SC;                              // reuse r2 as Z f32
    const size_t OFF_ZB    = OFF_SC    + 64ull * 1024 * 64 * 4;   // Z bf16, after ZF

    unsigned short* xb   = (unsigned short*)(ws + OFF_XB);
    unsigned short* wqk  = (unsigned short*)(ws + OFF_WQKV);
    unsigned short* wpr  = (unsigned short*)(ws + OFF_WPROJ);
    float*          qkvf = (float*)(ws + OFF_QKVF);
    unsigned short* qb   = (unsigned short*)(ws + OFF_QB);
    unsigned short* kb   = (unsigned short*)(ws + OFF_KB);
    unsigned short* vtb  = (unsigned short*)(ws + OFF_VTB);
    float*          S    = (float*)(ws + OFF_S);
    float*          Sc   = (float*)(ws + OFF_SC);
    unsigned short* P    = (unsigned short*)(ws + OFF_P);
    float*          Zf   = (float*)(ws + OFF_ZF);
    unsigned short* Zb   = (unsigned short*)(ws + OFF_ZB);

    const float scale = 0.125f;  // hd^-0.5 = 64^-0.5

    // 1) casts to bf16
    cast_f32_bf16<<<(8192 * 512 + 255) / 256, 256, 0, stream>>>(x, xb, 8192 * 512);
    cast_f32_bf16<<<(1536 * 512 + 255) / 256, 256, 0, stream>>>(qkv_w, wqk, 1536 * 512);
    cast_f32_bf16<<<(512 * 512 + 255) / 256, 256, 0, stream>>>(proj_w, wpr, 512 * 512);

    // 2) QKV GEMM: [8192,512] x [1536,512]^T -> [8192,1536] f32
    wmma_gemm<<<dim3(1536 / 64, 8192 / 256, 1), 256, 0, stream>>>(
        xb, wqk, qkvf, nullptr, nullptr, 8192, 1536, 512, 0, 0, 0);

    // 3) split/scale/transpose into per-head q, k, v^T
    repack_qkv<<<(1 << 22) / 256, 256, 0, stream>>>(qkvf, qb, kb, vtb, scale);

    // 4) scores S = (q*scale) k^T : 64 batched [1024,64]x[1024,64]^T
    wmma_gemm<<<dim3(1024 / 64, 1024 / 256, 64), 256, 0, stream>>>(
        qb, kb, S, nullptr, nullptr, 1024, 1024, 64,
        1024ll * 64, 1024ll * 64, 1024ll * 1024);

    // 5) conv3x3 + bias + BN + sigmoid on score map
    conv_bn_sigmoid<<<dim3(1024 / 256, 1024, 8), 256, 0, stream>>>(
        S, Sc, conv_w, conv_b, bn_g, bn_b, bn_m, bn_v);

    // 6) row softmax -> bf16 probabilities (overwrites region1)
    softmax_row<<<64 * 1024, 256, 0, stream>>>(Sc, P);

    // 7) Z = P V : 64 batched [1024,1024]x[64,1024]^T -> [1024,64] f32
    wmma_gemm<<<dim3(64 / 64, 1024 / 256, 64), 256, 0, stream>>>(
        P, vtb, Zf, nullptr, nullptr, 1024, 64, 1024,
        1024ll * 1024, 64ll * 1024, 1024ll * 64);

    // 8) merge heads -> [8192,512] bf16
    repack_z<<<(1 << 22) / 256, 256, 0, stream>>>(Zf, Zb);

    // 9) proj GEMM + bias + residual -> fp32 out
    wmma_gemm<<<dim3(512 / 64, 8192 / 256, 1), 256, 0, stream>>>(
        Zb, wpr, out, proj_b, x, 8192, 512, 512, 0, 0, 0);
}
